// SpGCN_36532991820141
// MI455X (gfx1250) — matrix operand order repaired and checked by reading, not compiled, not run on hardware
//
#include <hip/hip_runtime.h>
#include <hip/hip_bf16.h>

#define N_NODES   50000
#define N_EDGES   1600000
#define NFEAT     512
#define NHID      256
#define NCLASS    40
#define NCLASS_P  48   // padded to 3 x 16 WMMA tiles

typedef __attribute__((ext_vector_type(16))) __bf16 v16bf;
typedef __attribute__((ext_vector_type(8)))  __bf16 v8bf;
typedef __attribute__((ext_vector_type(4)))  __bf16 v4bf;
typedef __attribute__((ext_vector_type(8)))  float  v8f;

// ---------------------------------------------------------------------------
// f32 -> bf16 (RNE via compiler cast), 4 elements per thread
// ---------------------------------------------------------------------------
__global__ void k_f32_to_bf16(const float* __restrict__ in,
                              __bf16* __restrict__ out, int n) {
    int i = (blockIdx.x * blockDim.x + threadIdx.x) * 4;
    if (i + 3 < n) {
        float4 v = *(const float4*)(in + i);
        v4bf o;
        o[0] = (__bf16)v.x; o[1] = (__bf16)v.y;
        o[2] = (__bf16)v.z; o[3] = (__bf16)v.w;
        *(v4bf*)(out + i) = o;
    } else {
        for (int j = i; j < n; ++j) out[j] = (__bf16)in[j];
    }
}

// ---------------------------------------------------------------------------
// Transpose W[K,N] f32 -> WT[Npad,K] bf16, zero-padding rows n >= N
// ---------------------------------------------------------------------------
__global__ void k_transpose_bf16(const float* __restrict__ W,
                                 __bf16* __restrict__ WT,
                                 int K, int N, int Npad) {
    int idx = blockIdx.x * blockDim.x + threadIdx.x;
    if (idx >= Npad * K) return;
    int n = idx / K;
    int k = idx - n * K;
    float v = (n < N) ? W[k * N + n] : 0.0f;
    WT[(size_t)n * K + k] = (__bf16)v;
}

__global__ void k_zero_f32(float* __restrict__ p, int n) {
    int i = blockIdx.x * blockDim.x + threadIdx.x;
    if (i < n) p[i] = 0.0f;
}

// ---------------------------------------------------------------------------
// GEMM: C[M,N] = A[M,K] * B[K,N], with A bf16 row-major, BT bf16 = B^T
// (row n of BT is column n of B, contiguous in K). One wave per 16x16 tile,
// v_wmma_f32_16x16x32_bf16 over K/32 steps. M%16==0, N%16==0, K%32==0.
//
// A fragment (16-bit A 16x32, contiguous k order chosen here, same for B; not the ISA table order): lane l -> row m=l%16, group g=l/16,
//   half h: k = g*16 + h                -> one contiguous 16-half (32B) load.
// B fragment (16-bit B 32x16): lane l -> col n=l%16, group g=l/16,
//   half h: k = g*16 + h                -> one contiguous 16-half (32B) load.
// C/D: lane l, vgpr v -> row m = 8*(l/16)+v, col n = l%16.
// ---------------------------------------------------------------------------
__global__ void k_gemm_wmma_bf16(const __bf16* __restrict__ A,
                                 const __bf16* __restrict__ BT,
                                 float* __restrict__ C,
                                 int M, int N, int K) {
    int wave = (blockIdx.x * blockDim.x + threadIdx.x) >> 5;
    int lane = threadIdx.x & 31;
    int ntiles = N >> 4;
    int mt = wave / ntiles;
    int nt = wave - mt * ntiles;
    if (mt * 16 >= M) return;            // wave-uniform: EXEC stays all-ones

    int r = lane & 15;                   // row (A) / col (B) within tile
    int g = lane >> 4;                   // lane group

    const __bf16* arow = A  + (size_t)(mt * 16 + r) * K;
    const __bf16* brow = BT + (size_t)(nt * 16 + r) * K;

    v8f c = {0.f, 0.f, 0.f, 0.f, 0.f, 0.f, 0.f, 0.f};

    int kiters = K >> 5;
    for (int kk = 0; kk < kiters; ++kk) {
        int kb = kk * 32;
        v16bf a   = *(const v16bf*)(arow + kb + g * 16);
        v16bf b   = *(const v16bf*)(brow + kb + g * 16);
        c = __builtin_amdgcn_wmma_f32_16x16x32_bf16(
                /*neg_a=*/false, a, /*neg_b=*/false, b,
                /*c_mod=*/(short)0, c, /*reuse_a=*/false, /*reuse_b=*/false);
    }

    float* cbase = C + (size_t)(mt * 16 + g * 8) * N + nt * 16 + r;
#pragma unroll
    for (int v = 0; v < 8; ++v) cbase[(size_t)v * N] = c[v];
}

// ---------------------------------------------------------------------------
// SpMM scatter: one wave per edge. out[dst] += w * sup[src] over F features.
// Coalesced 128B gathers (stride-32 lanes), native f32 atomic scatter.
// support rows fit in the 192MB L2 -> gathers are mostly L2 hits.
// ---------------------------------------------------------------------------
__global__ void k_spmm_atomic(const float* __restrict__ sup,
                              const int* __restrict__ src,
                              const int* __restrict__ dst,
                              const float* __restrict__ w,
                              float* __restrict__ out,
                              int F, int ldsup, int ldout, int E) {
    int wave = (blockIdx.x * blockDim.x + threadIdx.x) >> 5;
    int lane = threadIdx.x & 31;
    if (wave >= E) return;
    int s = src[wave];
    int d = dst[wave];
    float wt = w[wave];
    const float* sp = sup + (size_t)s * ldsup;
    float*       op = out + (size_t)d * ldout;
    for (int f = lane; f < F; f += 32) {
        __hip_atomic_fetch_add(op + f, wt * sp[f],
                               __ATOMIC_RELAXED, __HIP_MEMORY_SCOPE_AGENT);
    }
}

// ---------------------------------------------------------------------------
// h = bf16(relu(in + b)), flat over n = rows*F elements
// ---------------------------------------------------------------------------
__global__ void k_bias_relu_bf16(const float* __restrict__ in,
                                 const float* __restrict__ b,
                                 __bf16* __restrict__ out,
                                 int F, int n) {
    int i = blockIdx.x * blockDim.x + threadIdx.x;
    if (i >= n) return;
    int c = i % F;
    float v = in[i] + b[c];
    out[i] = (__bf16)(v > 0.0f ? v : 0.0f);
}

// ---------------------------------------------------------------------------
// In-place row-wise log_softmax(io + b) over F classes; one wave per row.
// wave32 shuffle reductions.
// ---------------------------------------------------------------------------
__global__ void k_logsoftmax(float* __restrict__ io,
                             const float* __restrict__ b,
                             int nrows, int F) {
    int wave = (blockIdx.x * blockDim.x + threadIdx.x) >> 5;
    int lane = threadIdx.x & 31;
    if (wave >= nrows) return;
    float* row = io + (size_t)wave * F;
    int c0 = lane, c1 = lane + 32;
    float v0 = -__builtin_inff(), v1 = -__builtin_inff();
    if (c0 < F) v0 = row[c0] + b[c0];
    if (c1 < F) v1 = row[c1] + b[c1];
    float m = fmaxf(v0, v1);
#pragma unroll
    for (int o = 16; o > 0; o >>= 1) m = fmaxf(m, __shfl_xor(m, o, 32));
    float s = 0.0f;
    if (c0 < F) s += expf(v0 - m);
    if (c1 < F) s += expf(v1 - m);
#pragma unroll
    for (int o = 16; o > 0; o >>= 1) s += __shfl_xor(s, o, 32);
    float lse = logf(s) + m;
    if (c0 < F) row[c0] = v0 - lse;
    if (c1 < F) row[c1] = v1 - lse;
}

// ---------------------------------------------------------------------------
// Launch pipeline
// ---------------------------------------------------------------------------
extern "C" void kernel_launch(void* const* d_in, const int* in_sizes, int n_in,
                              void* d_out, int out_size, void* d_ws, size_t ws_size,
                              hipStream_t stream) {
    const float* x  = (const float*)d_in[0];
    const int*   ei = (const int*)  d_in[1];
    const float* ew = (const float*)d_in[2];
    const float* W1 = (const float*)d_in[3];
    const float* b1 = (const float*)d_in[4];
    const float* W2 = (const float*)d_in[5];
    const float* b2 = (const float*)d_in[6];
    float* out = (float*)d_out;

    const int* src = ei;            // edge_index[0]
    const int* dst = ei + N_EDGES;  // edge_index[1]

    // Workspace layout (all offsets 256B-aligned)
    char* ws = (char*)d_ws;
    size_t off = 0;
    __bf16* xbf  = (__bf16*)(ws + off); off += (size_t)N_NODES * NFEAT * 2;   // 51.2 MB
    __bf16* w1t  = (__bf16*)(ws + off); off += (size_t)NHID * NFEAT * 2;      // 0.26 MB
    __bf16* w2t  = (__bf16*)(ws + off); off += (size_t)NCLASS_P * NHID * 2;   // 24 KB
    float*  sup1 = (float*) (ws + off); off += (size_t)N_NODES * NHID * 4;    // 51.2 MB
    float*  out1 = (float*) (ws + off); off += (size_t)N_NODES * NHID * 4;    // 51.2 MB
    float*  sup2 = (float*) (ws + off); off += (size_t)N_NODES * NCLASS_P * 4;// 9.6 MB
    __bf16* hbf  = xbf;  // reuse x_bf16 region after GEMM1 consumed it

    // 1) x -> bf16
    {
        int n = N_NODES * NFEAT;
        int th = (n + 3) / 4;
        k_f32_to_bf16<<<(th + 255) / 256, 256, 0, stream>>>(x, xbf, n);
    }
    // 2) W1 -> W1T bf16 ; W2 -> W2T bf16 (padded 40->48 rows)
    {
        int n1 = NHID * NFEAT;
        k_transpose_bf16<<<(n1 + 255) / 256, 256, 0, stream>>>(W1, w1t, NFEAT, NHID, NHID);
        int n2 = NCLASS_P * NHID;
        k_transpose_bf16<<<(n2 + 255) / 256, 256, 0, stream>>>(W2, w2t, NHID, NCLASS, NCLASS_P);
    }
    // 3) support1 = x @ W1   (WMMA bf16 -> f32)
    {
        int waves = (N_NODES / 16) * (NHID / 16);     // 3125 * 16
        int blocks = (waves * 32 + 255) / 256;
        k_gemm_wmma_bf16<<<blocks, 256, 0, stream>>>(xbf, w1t, sup1, N_NODES, NHID, NFEAT);
    }
    // 4) out1 = scatter_add(w * support1[src]) ; zero first
    {
        int n = N_NODES * NHID;
        k_zero_f32<<<(n + 255) / 256, 256, 0, stream>>>(out1, n);
        int blocks = (N_EDGES * 32 + 255) / 256;
        k_spmm_atomic<<<blocks, 256, 0, stream>>>(sup1, src, dst, ew, out1,
                                                  NHID, NHID, NHID, N_EDGES);
    }
    // 5) h = bf16(relu(out1 + b1))
    {
        int n = N_NODES * NHID;
        k_bias_relu_bf16<<<(n + 255) / 256, 256, 0, stream>>>(out1, b1, hbf, NHID, n);
    }
    // 6) support2 = h @ W2   (padded N=48)
    {
        int waves = (N_NODES / 16) * (NCLASS_P / 16); // 3125 * 3
        int blocks = (waves * 32 + 255) / 256;
        k_gemm_wmma_bf16<<<blocks, 256, 0, stream>>>(hbf, w2t, sup2, N_NODES, NCLASS_P, NHID);
    }
    // 7) d_out = scatter_add(w * support2[src]) over 40 classes; zero first
    {
        int n = N_NODES * NCLASS;
        k_zero_f32<<<(n + 255) / 256, 256, 0, stream>>>(out, n);
        int blocks = (N_EDGES * 32 + 255) / 256;
        k_spmm_atomic<<<blocks, 256, 0, stream>>>(sup2, src, dst, ew, out,
                                                  NCLASS, NCLASS_P, NCLASS, N_EDGES);
    }
    // 8) in-place log_softmax(d_out + b2)
    {
        int blocks = (N_NODES * 32 + 255) / 256;
        k_logsoftmax<<<blocks, 256, 0, stream>>>(out, b2, N_NODES, NCLASS);
    }
    (void)in_sizes; (void)n_in; (void)out_size; (void)ws_size;
}
